// PartialFC_V2_22582938043069
// MI455X (gfx1250) — matrix operand order repaired and checked by reading, not compiled, not run on hardware
//
#include <hip/hip_runtime.h>
#include <hip/hip_bf16.h>

// ---------------- problem constants ----------------
#define B_ROWS   512
#define D_DIM    512
#define C_CLS    200000
#define SCALE_S  64.0f
#define COS_M    0.8775825618903728f   // cos(0.5)
#define SIN_M    0.4794255386042030f   // sin(0.5)
#define EPS_N    1e-12f
#define NEG_BIG  -3.0e38f
#define CLS_PER_BLK 512
#define NBLK_C   391                   // ceil(200000/512)
#define WT_LD    520                   // 512 + 8 bf16 pad (bank-conflict-free)

// ---------------- vector types ----------------
typedef __bf16 v16bf __attribute__((ext_vector_type(16)));
typedef float  v8f   __attribute__((ext_vector_type(8)));
typedef float  f32x4 __attribute__((ext_vector_type(4)));
typedef unsigned int u32x4 __attribute__((ext_vector_type(4)));
typedef unsigned int u32x2 __attribute__((ext_vector_type(2)));

union BFrag { v16bf v; u32x4 q[2]; };

__device__ __forceinline__ unsigned short f2bf(float f) {
    unsigned u = __float_as_uint(f);
    unsigned r = u + 0x7FFFu + ((u >> 16) & 1u);   // RNE
    return (unsigned short)(r >> 16);
}

__device__ __forceinline__ void lse_merge(float& m1, float& s1, float m2, float s2) {
    float nm = fmaxf(m1, m2);
    s1 = s1 * __expf(m1 - nm) + s2 * __expf(m2 - nm);
    m1 = nm;
}

// ============ kernel 1: normalize embeddings -> bf16 A, store 1/||e|| ============
__global__ void k_norm_embed(const float* __restrict__ E,
                             unsigned short* __restrict__ Abf,
                             float* __restrict__ inv_e) {
    const int i = blockIdx.x, t = threadIdx.x;          // 512 blocks x 128 thr
    f32x4 x = ((const f32x4*)(E + (size_t)i * D_DIM))[t];
    float ss = x[0]*x[0] + x[1]*x[1] + x[2]*x[2] + x[3]*x[3];
    #pragma unroll
    for (int s2 = 16; s2 >= 1; s2 >>= 1) ss += __shfl_xor(ss, s2);
    __shared__ float sm[4];
    if ((t & 31) == 0) sm[t >> 5] = ss;
    __syncthreads();
    float tot = sm[0] + sm[1] + sm[2] + sm[3];
    float inv = 1.0f / fmaxf(sqrtf(tot), EPS_N);
    unsigned p0 = (unsigned)f2bf(x[0]*inv) | ((unsigned)f2bf(x[1]*inv) << 16);
    unsigned p1 = (unsigned)f2bf(x[2]*inv) | ((unsigned)f2bf(x[3]*inv) << 16);
    *(u32x2*)(Abf + (size_t)i * D_DIM + t * 4) = (u32x2){p0, p1};
    if (t == 0) inv_e[i] = inv;
}

// ============ kernel 2: f32 gather of target cosine ============
__global__ void k_gather(const float* __restrict__ E, const float* __restrict__ W,
                         const int* __restrict__ labels, const float* __restrict__ inv_e,
                         float* __restrict__ tgt) {
    const int i = blockIdx.x, t = threadIdx.x;          // 512 blocks x 128 thr
    int c = labels[i];
    long long c0 = (c < 0) ? 0 : c;
    f32x4 e  = ((const f32x4*)(E + (size_t)i  * D_DIM))[t];
    f32x4 wv = ((const f32x4*)(W + (size_t)c0 * D_DIM))[t];
    float d  = e[0]*wv[0] + e[1]*wv[1] + e[2]*wv[2] + e[3]*wv[3];
    float ws = wv[0]*wv[0] + wv[1]*wv[1] + wv[2]*wv[2] + wv[3]*wv[3];
    #pragma unroll
    for (int s2 = 16; s2 >= 1; s2 >>= 1) { d += __shfl_xor(d, s2); ws += __shfl_xor(ws, s2); }
    __shared__ float sd[4], sw[4];
    if ((t & 31) == 0) { sd[t >> 5] = d; sw[t >> 5] = ws; }
    __syncthreads();
    if (t == 0) {
        float dt = sd[0]+sd[1]+sd[2]+sd[3];
        float wt = sw[0]+sw[1]+sw[2]+sw[3];
        tgt[i] = dt * inv_e[i] / fmaxf(sqrtf(wt), EPS_N);
    }
}

// ============ kernel 3: streaming bf16 WMMA GEMM + online logsumexp ============
__global__ __launch_bounds__(256)
void k_gemm(const float* __restrict__ W, const unsigned short* __restrict__ Abf,
            float* __restrict__ gpart) {
    __shared__ unsigned short Wt[128 * WT_LD];   // normalized bf16 weight tile (133 KB)
    __shared__ float Sf[8 * D_DIM];              // async f32 staging (16 KB)
    __shared__ float Sm[8][16], Ss[8][16];       // per-wave per-row partials
    __shared__ float Bm[B_ROWS], Bs[B_ROWS];     // block-level running (max,sum)

    const int tid = threadIdx.x, w = tid >> 5, lane = tid & 31;
    const int blk = blockIdx.x;
    const long long cb = (long long)blk * CLS_PER_BLK;

    for (int r = tid; r < B_ROWS; r += 256) { Bm[r] = NEG_BIG; Bs[r] = 0.0f; }
    __syncthreads();

    const unsigned sf_base = (unsigned)(unsigned long long)(void*)Sf;

    for (int ch = 0; ch < 4; ++ch) {
        const long long chbase = cb + ch * 128;

        // ---- stage + row-normalize 128 weight rows (f32 -> bf16 LDS) ----
        for (int sg = 0; sg < 16; ++sg) {
            const long long rbase = chbase + sg * 8;
            #pragma unroll
            for (int it = 0; it < 4; ++it) {
                int idx = w * 128 + it * 32 + lane;         // 1024 x 16B chunks
                int r8  = idx >> 7, col = idx & 127;        // col in float4 units
                long long cr = rbase + r8; if (cr > C_CLS - 1) cr = C_CLS - 1;
                unsigned long long ga = (unsigned long long)(W + cr * D_DIM + col * 4);
                unsigned la = sf_base + (unsigned)((r8 * D_DIM + col * 4) * 4);
                asm volatile("global_load_async_to_lds_b128 %0, %1, off"
                             :: "v"(la), "v"(ga) : "memory");
            }
            asm volatile("s_wait_asynccnt 0" ::: "memory");
            __syncthreads();
            {   // wave w normalizes staged row w
                float v[16]; float ss = 0.0f;
                #pragma unroll
                for (int it = 0; it < 4; ++it) {
                    f32x4 q = *(const f32x4*)(Sf + w * D_DIM + it * 128 + lane * 4);
                    v[it*4+0]=q[0]; v[it*4+1]=q[1]; v[it*4+2]=q[2]; v[it*4+3]=q[3];
                    ss += q[0]*q[0] + q[1]*q[1] + q[2]*q[2] + q[3]*q[3];
                }
                #pragma unroll
                for (int s2 = 16; s2 >= 1; s2 >>= 1) ss += __shfl_xor(ss, s2);
                float inv = 1.0f / fmaxf(sqrtf(ss), EPS_N);
                int wrow = sg * 8 + w;
                #pragma unroll
                for (int it = 0; it < 4; ++it) {
                    unsigned p0 = (unsigned)f2bf(v[it*4+0]*inv) | ((unsigned)f2bf(v[it*4+1]*inv) << 16);
                    unsigned p1 = (unsigned)f2bf(v[it*4+2]*inv) | ((unsigned)f2bf(v[it*4+3]*inv) << 16);
                    *(u32x2*)(Wt + wrow * WT_LD + it * 128 + lane * 4) = (u32x2){p0, p1};
                }
            }
            __syncthreads();
        }

        // ---- compute: each wave = 16 classes, all 512 rows ----
        const long long clsbase = chbase + w * 16;
        const bool nv = (clsbase + (lane & 15)) < C_CLS;
        const int m = lane & 15, h = lane >> 4;
        const unsigned short* brow = Wt + (w * 16 + m) * WT_LD;

        for (int rt = 0; rt < 32; ++rt) {
            v8f acc = {0.f,0.f,0.f,0.f,0.f,0.f,0.f,0.f};
            const unsigned short* arow = Abf + (size_t)(rt * 16 + m) * D_DIM;
            #pragma unroll
            for (int kt = 0; kt < 16; ++kt) {
                BFrag a, b;
                int ka = kt * 32 + h * 8;       // A 16x32 bf16 lane layout
                a.q[0] = *(const u32x4*)(arow + ka);
                a.q[1] = *(const u32x4*)(arow + ka + 16);
                int kb = kt * 32 + h * 8;       // B 32x16 bf16 lane layout
                b.q[0] = *(const u32x4*)(brow + kb);
                b.q[1] = *(const u32x4*)(brow + kb + 16);
                acc = __builtin_amdgcn_wmma_f32_16x16x32_bf16(
                        false, a.v, false, b.v, (short)0, acc, false, false);
            }
            // per-row (over 16 columns) max / exp-sum
            float rmax[8], rsum[8];
            #pragma unroll
            for (int r = 0; r < 8; ++r) {
                float x  = nv ? SCALE_S * acc[r] : NEG_BIG;
                float mx = x;
                mx = fmaxf(mx, __shfl_xor(mx, 1));
                mx = fmaxf(mx, __shfl_xor(mx, 2));
                mx = fmaxf(mx, __shfl_xor(mx, 4));
                mx = fmaxf(mx, __shfl_xor(mx, 8));
                float e = nv ? __expf(x - mx) : 0.0f;
                e += __shfl_xor(e, 1); e += __shfl_xor(e, 2);
                e += __shfl_xor(e, 4); e += __shfl_xor(e, 8);
                rmax[r] = mx; rsum[r] = e;
            }
            #pragma unroll
            for (int r = 0; r < 8; ++r) {
                if ((lane & 15) == r) { Sm[w][h*8+r] = rmax[r]; Ss[w][h*8+r] = rsum[r]; }
            }
            __syncthreads();
            if (tid < 16) {
                float m2 = NEG_BIG, s2 = 0.0f;
                #pragma unroll
                for (int q = 0; q < 8; ++q) lse_merge(m2, s2, Sm[q][tid], Ss[q][tid]);
                lse_merge(Bm[rt*16 + tid], Bs[rt*16 + tid], m2, s2);
            }
            __syncthreads();
        }
    }
    for (int r = tid; r < B_ROWS; r += 256) {
        size_t o = ((size_t)blk * B_ROWS + r) * 2;
        gpart[o]     = Bm[r];
        gpart[o + 1] = Bs[r];
    }
}

// ============ kernel 4: per-row merge + arcface-corrected NLL ============
__global__ void k_finalize(const float* __restrict__ gpart, const float* __restrict__ tgt,
                           const int* __restrict__ labels,
                           float* __restrict__ row_loss, float* __restrict__ row_valid) {
    const int row = blockIdx.x, t = threadIdx.x;        // 512 blocks x 256 thr
    float m = NEG_BIG, s = 0.0f;
    for (int b = t; b < NBLK_C; b += 256) {
        size_t o = ((size_t)b * B_ROWS + row) * 2;
        lse_merge(m, s, gpart[o], gpart[o + 1]);
    }
    __shared__ float Rm[256], Rs[256];
    Rm[t] = m; Rs[t] = s;
    __syncthreads();
    for (int st = 128; st > 0; st >>= 1) {
        if (t < st) lse_merge(Rm[t], Rs[t], Rm[t + st], Rs[t + st]);
        __syncthreads();
    }
    if (t == 0) {
        float M = Rm[0], S = Rs[0];
        int   lab = labels[row];
        bool  valid = lab >= 0;
        float tg = tgt[row];
        float tc = fminf(fmaxf(tg, -1.0f + 1e-7f), 1.0f - 1e-7f);
        float nt = tc * COS_M - sqrtf(fmaxf(1.0f - tc * tc, 0.0f)) * SIN_M;
        float applied = valid ? nt : tg;
        float S2 = S - __expf(SCALE_S * tg - M) + __expf(SCALE_S * applied - M);
        S2 = fmaxf(S2, 1e-30f);
        float nll = M + logf(S2) - SCALE_S * applied;
        row_loss[row]  = valid ? nll : 0.0f;
        row_valid[row] = valid ? 1.0f : 0.0f;
    }
}

// ============ kernel 5: mean over valid rows ============
__global__ void k_final(const float* __restrict__ row_loss, const float* __restrict__ row_valid,
                        float* __restrict__ out) {
    __shared__ float sl[512], sv[512];
    const int t = threadIdx.x;
    sl[t] = row_loss[t]; sv[t] = row_valid[t];
    __syncthreads();
    for (int st = 256; st > 0; st >>= 1) {
        if (t < st) { sl[t] += sl[t + st]; sv[t] += sv[t + st]; }
        __syncthreads();
    }
    if (t == 0) out[0] = sl[0] / fmaxf(sv[0], 1.0f);
}

// ============ launcher ============
extern "C" void kernel_launch(void* const* d_in, const int* in_sizes, int n_in,
                              void* d_out, int out_size, void* d_ws, size_t ws_size,
                              hipStream_t stream) {
    const float* E      = (const float*)d_in[0];   // [512,512]
    const int*   labels = (const int*)d_in[1];     // [512]
    const float* W      = (const float*)d_in[2];   // [200000,512]
    float* out = (float*)d_out;

    char* ws = (char*)d_ws;
    unsigned short* Abf   = (unsigned short*)(ws);                         // 524288 B
    float* inv_e          = (float*)(ws + 524288);                         // 2048 B
    float* tgt            = (float*)(ws + 526336);                         // 2048 B
    float* gpart          = (float*)(ws + 528384);                         // 391*512*2*4 B
    float* row_loss       = (float*)(ws + 528384 + (size_t)NBLK_C*B_ROWS*2*4);
    float* row_valid      = row_loss + B_ROWS;

    k_norm_embed<<<B_ROWS, 128, 0, stream>>>(E, Abf, inv_e);
    k_gather    <<<B_ROWS, 128, 0, stream>>>(E, W, labels, inv_e, tgt);
    k_gemm      <<<NBLK_C, 256, 0, stream>>>(W, Abf, gpart);
    k_finalize  <<<B_ROWS, 256, 0, stream>>>(gpart, tgt, labels, row_loss, row_valid);
    k_final     <<<1, 512, 0, stream>>>(row_loss, row_valid, out);
}
